// MoEPredictor_89893665505573
// MI455X (gfx1250) — matrix operand order repaired and checked by reading, not compiled, not run on hardware
//
#include <hip/hip_runtime.h>

#define Nn 4096
#define Dd 256
#define Hh 1024
#define Ee 8

#define XS_STRIDE  264   // 256 + 8 halves pad (528B rows, 16B aligned)
#define HS_STRIDE  72    // 64 + 8 halves pad

#define MAX_TILES 128    // ceil(N/32) worst case
#define SMEM_HALVES (32*XS_STRIDE + 32*HS_STRIDE)
#define SMEM_BYTES  (SMEM_HALVES * 2)

// workspace layout (bytes)
#define WS_COUNTS 0
#define WS_LIST   256                       // E*N ints  = 131072 B
#define WS_XH     131584                    // N*D f16   = 2 MB   (16B aligned)
#define WS_W1T    (WS_XH  + Nn*Dd*2)        // E*H*D f16 = 4 MB   [e][h][d]
#define WS_W2T    (WS_W1T + Ee*Dd*Hh*2)     // E*D*H f16 = 4 MB   [e][d][h]

typedef __attribute__((ext_vector_type(16))) _Float16 v16h;
typedef __attribute__((ext_vector_type(8)))  _Float16 v8h;
typedef __attribute__((ext_vector_type(4)))  _Float16 v4h;
typedef __attribute__((ext_vector_type(8)))  float    v8f;
typedef __attribute__((ext_vector_type(4)))  unsigned int u32x4;
typedef __attribute__((ext_vector_type(8)))  int      i32x8;
typedef __attribute__((ext_vector_type(4)))  int      i32x4;

static __device__ __forceinline__ v8f wmma_f16(v16h a, v16h b, v8f c) {
  return __builtin_amdgcn_wmma_f32_16x16x32_f16(
      false, a, false, b, (short)0, c, false, false);
}

// A fragment (16x32 f16) from LDS: lanes 0-15 row M hold K 0-7 & 16-23,
// lanes 16-31 hold K 8-15 & 24-31; two 16B ds reads.
static __device__ __forceinline__ v16h load_a_frag(const _Float16* base, int stride,
                                                   int row, int kbase, int lane) {
  int k0 = kbase + ((lane >> 4) & 1) * 8;
  const _Float16* p = base + row * stride + k0;
  v8h lo = *(const v8h*)(p);
  v8h hi = *(const v8h*)(p + 16);
  v16h a;
#pragma unroll
  for (int i = 0; i < 8; ++i) { a[i] = lo[i]; a[i + 8] = hi[i]; }
  return a;
}

// two 8-half runs at p and p+16 (global, 16B aligned) -> B fragment half-row
static __device__ __forceinline__ v16h load_16h(const _Float16* p) {
  v8h lo = *(const v8h*)(p);
  v8h hi = *(const v8h*)(p + 16);
  v16h b;
#pragma unroll
  for (int i = 0; i < 8; ++i) { b[i] = lo[i]; b[i + 8] = hi[i]; }
  return b;
}

// ---------------------------------------------------------------------------
// TDM gather: pull 16 token rows (256 f16 each) from xh into LDS with a
// padded stride of 528B.  One descriptor, issued once per wave.
// ---------------------------------------------------------------------------
static __device__ __forceinline__ void tdm_gather16(const _Float16* xh,
                                                    unsigned int lds_byte_addr,
                                                    const int* toks, int base) {
  unsigned long long ga = (unsigned long long)(uintptr_t)xh;
  // group0: count=1, gather_mode=1 (16-bit idx), lds_addr, global_addr, type=2
  u32x4 g0 = { 0x80000001u,
               lds_byte_addr,
               (unsigned int)ga,
               (unsigned int)(ga >> 32) | 0x80000000u };
  // group1: data_size=2B, pad_enable, pad_interval=128DW, pad_amount=4DW,
  //         tensor_dim0=256, tensor_dim1=4096, tile_dim0=256, tile_dim1=16,
  //         tensor_dim0_stride=256
  i32x8 g1 = { (int)0x07910000u,
               (int)(256u << 16),      // tensor_dim0 low16 in [63:48]
               (int)(4096u << 16),     // tensor_dim1 low16 in [95:80]
               (int)(256u << 16),      // tile_dim0 in [127:112]
               16,                     // tile_dim1 = #indices
               256,                    // tensor_dim0_stride low32
               0, 0 };
  int w[8];
#pragma unroll
  for (int i = 0; i < 8; ++i) {
    int t0 = toks[base + 2*i];     if (t0 < 0) t0 = 0;
    int t1 = toks[base + 2*i + 1]; if (t1 < 0) t1 = 0;
    w[i] = __builtin_amdgcn_readfirstlane((t0 & 0xFFFF) | (t1 << 16));
  }
  i32x4 g2 = { w[0], w[1], w[2], w[3] };
  i32x4 g3 = { w[4], w[5], w[6], w[7] };
  i32x8 g4 = { 0, 0, 0, 0, 0, 0, 0, 0 };   // reserved group (6-arg toolchain)
  __builtin_amdgcn_tensor_load_to_lds(g0, g1, g2, g3, g4, 0);
}

// ---------------------------------------------------------------------------
// Kernel 1: zero out + counters, convert x -> f16
// ---------------------------------------------------------------------------
__global__ void moe_init(const float* __restrict__ x, _Float16* __restrict__ xh,
                         float* __restrict__ out, int* __restrict__ counts) {
  int idx = blockIdx.x * blockDim.x + threadIdx.x;   // 262144 = N*D/4
  float4 z = {0.f, 0.f, 0.f, 0.f};
  ((float4*)out)[idx] = z;
  float4 v = ((const float4*)x)[idx];
  v4h hv = { (_Float16)v.x, (_Float16)v.y, (_Float16)v.z, (_Float16)v.w };
  *(v4h*)(xh + (size_t)idx * 4) = hv;
  if (idx < Ee) counts[idx] = 0;
}

// ---------------------------------------------------------------------------
// Kernel 2: f32 [e][rows][cols] -> f16 [e][cols][rows], LDS-tiled transpose
// ---------------------------------------------------------------------------
__global__ void cvt_transpose(const float* __restrict__ in, _Float16* __restrict__ outp,
                              int rows, int cols) {
  __shared__ float tile[32][33];
  int e  = blockIdx.z;
  int r0 = blockIdx.y * 32;
  int c0 = blockIdx.x * 32;
  int tx = threadIdx.x & 31, ty = threadIdx.x >> 5;   // 32 x 8
  const float* ip = in + (size_t)e * rows * cols;
  _Float16* op = outp + (size_t)e * rows * cols;
#pragma unroll
  for (int s = 0; s < 32; s += 8)
    tile[ty + s][tx] = ip[(size_t)(r0 + ty + s) * cols + c0 + tx];
  __syncthreads();
#pragma unroll
  for (int s = 0; s < 32; s += 8)
    op[(size_t)(c0 + ty + s) * rows + r0 + tx] = (_Float16)tile[tx][ty + s];
}

// ---------------------------------------------------------------------------
// Kernel 3: gate + top-2 + scatter into per-expert token lists (wave per token)
// ---------------------------------------------------------------------------
__global__ void moe_gate(const float* __restrict__ x, const float* __restrict__ gW,
                         const float* __restrict__ gb, const float* __restrict__ tptr,
                         int* __restrict__ counts, int* __restrict__ list) {
  const int lane  = threadIdx.x & 31;
  const int wave  = threadIdx.x >> 5;
  const int token = blockIdx.x * 8 + wave;
  const float temp = *tptr;

  const float4* xp = (const float4*)(x + token * Dd + lane * 8);
  float4 xa = xp[0], xb = xp[1];

  float s[Ee];
#pragma unroll
  for (int e = 0; e < Ee; ++e) {
    const float4* wp = (const float4*)(gW + e * Dd + lane * 8);
    float4 wa = wp[0], wb = wp[1];
    float p = xa.x * wa.x + xa.y * wa.y + xa.z * wa.z + xa.w * wa.w
            + xb.x * wb.x + xb.y * wb.y + xb.z * wb.z + xb.w * wb.w;
#pragma unroll
    for (int off = 16; off > 0; off >>= 1) p += __shfl_xor(p, off, 32);
    s[e] = (p + gb[e]) / temp;
  }

  if (lane == 0) {
    int i1 = 0; float v1 = s[0];
#pragma unroll
    for (int e = 1; e < Ee; ++e) if (s[e] > v1) { v1 = s[e]; i1 = e; }
    int i2 = -1; float v2 = -3.402823466e38f;
#pragma unroll
    for (int e = 0; e < Ee; ++e) if (e != i1 && s[e] > v2) { v2 = s[e]; i2 = e; }
    int p1 = atomicAdd(&counts[i1], 1);
    list[i1 * Nn + p1] = token;
    int p2 = atomicAdd(&counts[i2], 1);
    list[i2 * Nn + p2] = token;
  }
}

// ---------------------------------------------------------------------------
// Kernel 4: grouped expert FFN.  X tile gathered into LDS via TDM; WMMA B
// fragments read straight from pre-transposed f16 weights in global (L2).
// ---------------------------------------------------------------------------
__global__ void moe_expert(const _Float16* __restrict__ xh,
                           const _Float16* __restrict__ W1T,  // [e][h][d]
                           const float*    __restrict__ b1g,
                           const _Float16* __restrict__ W2T,  // [e][d][h]
                           const float*    __restrict__ b2g,
                           const int* __restrict__ counts, const int* __restrict__ list,
                           float* __restrict__ out) {
  extern __shared__ _Float16 smem[];
  _Float16* Xs = smem;                       // [32][264]
  _Float16* Hs = smem + 32 * XS_STRIDE;      // [32][72]
  __shared__ int toks[32];

  const int e    = blockIdx.x >> 7;
  const int tile = blockIdx.x & (MAX_TILES - 1);
  const int cnt  = counts[e];
  const int row0 = tile * 32;
  if (row0 >= cnt) return;

  const int tid  = threadIdx.x;
  const int lane = tid & 31;
  const int wave = tid >> 5;
  const int l15  = lane & 15;

  if (tid < 32) {
    int r = row0 + tid;
    toks[tid] = (r < cnt) ? list[e * Nn + r] : -1;
  }
  __syncthreads();

  // ---- TDM gather: 32 token rows -> Xs (two 16-row descriptors), wave 0 ----
  if (wave == 0) {
    unsigned int xs0 = (unsigned int)(uintptr_t)Xs;
    tdm_gather16(xh, xs0, toks, 0);
    tdm_gather16(xh, xs0 + 16u * XS_STRIDE * 2u, toks, 16);
    __builtin_amdgcn_s_wait_tensorcnt(0);
  }
  __syncthreads();

  const _Float16* W1e = W1T + (size_t)e * Hh * Dd;
  const _Float16* W2e = W2T + (size_t)e * Dd * Hh;

  // persistent y tiles: wave owns (mb 0..1) x (db = wave*2 + j)
  v8f yacc[2][2];
#pragma unroll
  for (int a = 0; a < 2; ++a)
#pragma unroll
    for (int b = 0; b < 2; ++b) yacc[a][b] = {};

  const int mb1 = wave >> 2;                 // stage-1 tile coords
  const int jb1 = wave & 3;
  const int khalf = ((lane >> 4) & 1) * 8;   // B half-row select

  for (int hc = 0; hc < Hh; hc += 64) {
    __syncthreads();   // previous chunk's Hs readers done

    // ---- stage 1: h = relu(X @ W1chunk + b1); one 16x16 tile per wave ----
    {
      const _Float16* w1row = W1e + (size_t)(hc + jb1 * 16 + l15) * Dd + khalf;
      if (hc + 64 < Hh) __builtin_prefetch(w1row + 64 * Dd, 0, 1);
      v8f acc = {};
#pragma unroll
      for (int kk = 0; kk < 8; ++kk) {
        v16h a = load_a_frag(Xs, XS_STRIDE, mb1 * 16 + l15, kk * 32, lane);
        v16h b = load_16h(w1row + kk * 32);
        acc = wmma_f16(a, b, acc);
      }
      float bias = b1g[e * Hh + hc + jb1 * 16 + l15];
      int mrow = 8 * (lane >> 4);
#pragma unroll
      for (int v = 0; v < 8; ++v) {
        float h = acc[v] + bias;
        h = h > 0.f ? h : 0.f;
        Hs[(mb1 * 16 + mrow + v) * HS_STRIDE + jb1 * 16 + l15] = (_Float16)h;
      }
    }
    __syncthreads();

    // ---- stage 2: y += h @ W2chunk (4 tiles per wave, K = 64) ----
    v16h bfr[2][2], afr[2][2];
#pragma unroll
    for (int j = 0; j < 2; ++j) {
      const _Float16* w2row = W2e + (size_t)((wave * 2 + j) * 16 + l15) * Hh + hc + khalf;
      if (j == 0 && hc + 64 < Hh) __builtin_prefetch(w2row + 64, 0, 1);
#pragma unroll
      for (int kk = 0; kk < 2; ++kk) bfr[j][kk] = load_16h(w2row + kk * 32);
    }
#pragma unroll
    for (int mb = 0; mb < 2; ++mb)
#pragma unroll
      for (int kk = 0; kk < 2; ++kk)
        afr[mb][kk] = load_a_frag(Hs, HS_STRIDE, mb * 16 + l15, kk * 32, lane);
#pragma unroll
    for (int j = 0; j < 2; ++j)
#pragma unroll
      for (int mb = 0; mb < 2; ++mb)
#pragma unroll
        for (int kk = 0; kk < 2; ++kk)
          yacc[mb][j] = wmma_f16(afr[mb][kk], bfr[j][kk], yacc[mb][j]);
  }

  // ---- epilogue: +b2, accumulate (exactly 2 commutative f32 adds/element) ----
#pragma unroll
  for (int j = 0; j < 2; ++j) {
    int dcol = (wave * 2 + j) * 16 + l15;
    float bias = b2g[e * Dd + dcol];
#pragma unroll
    for (int mb = 0; mb < 2; ++mb) {
      int mbase = mb * 16 + 8 * (lane >> 4);
#pragma unroll
      for (int v = 0; v < 8; ++v) {
        int tok = toks[mbase + v];
        if (tok >= 0) {
          unsafeAtomicAdd(out + tok * Dd + dcol, yacc[mb][j][v] + bias);
        }
      }
    }
  }
}

// ---------------------------------------------------------------------------
extern "C" void kernel_launch(void* const* d_in, const int* in_sizes, int n_in,
                              void* d_out, int out_size, void* d_ws, size_t ws_size,
                              hipStream_t stream) {
  (void)in_sizes; (void)n_in; (void)out_size; (void)ws_size;
  const float* x    = (const float*)d_in[0];
  const float* gW   = (const float*)d_in[1];
  const float* gb   = (const float*)d_in[2];
  const float* W1   = (const float*)d_in[3];
  const float* b1   = (const float*)d_in[4];
  const float* W2   = (const float*)d_in[5];
  const float* b2   = (const float*)d_in[6];
  const float* temp = (const float*)d_in[7];
  float* out = (float*)d_out;

  char* ws = (char*)d_ws;
  int*       counts = (int*)(ws + WS_COUNTS);
  int*       list   = (int*)(ws + WS_LIST);
  _Float16*  xh     = (_Float16*)(ws + WS_XH);
  _Float16*  W1Tp   = (_Float16*)(ws + WS_W1T);
  _Float16*  W2Tp   = (_Float16*)(ws + WS_W2T);

  moe_init<<<dim3((Nn * Dd / 4) / 256), dim3(256), 0, stream>>>(x, xh, out, counts);
  // W1 [e][D][H] -> W1T [e][H][D]
  cvt_transpose<<<dim3(Hh / 32, Dd / 32, Ee), dim3(256), 0, stream>>>(W1, W1Tp, Dd, Hh);
  // W2 [e][H][D] -> W2T [e][D][H]
  cvt_transpose<<<dim3(Dd / 32, Hh / 32, Ee), dim3(256), 0, stream>>>(W2, W2Tp, Hh, Dd);
  moe_gate<<<dim3(Nn / 8), dim3(256), 0, stream>>>(x, gW, gb, temp, counts, list);
  moe_expert<<<dim3(Ee * MAX_TILES), dim3(256), SMEM_BYTES, stream>>>(
      xh, W1Tp, b1, W2Tp, b2, counts, list, out);
}
